// DGCNN_gpvn_scene_74071005987491
// MI455X (gfx1250) — matrix operand order repaired and checked by reading, compiled, not run on hardware
//
#include <hip/hip_runtime.h>

// DGCNN forward for MI455X (gfx1250, wave32, WMMA bf16).
// All tensors that feed WMMA are pre-packed bf16 in point-major rows so every
// fragment is two global_load_b128 ops. Edge-feature subtraction is folded
// into the weights (W·[nbr-ctr;ctr] = Wa·nbr + (Wb-Wa)·ctr), so gather convs
// do pure row gathers. Each wave register-blocks 4 column tiles (16x64 out),
// with all 4 B-fragments preloaded before the WMMA burst for ILP.
// top-k stages pd rows into LDS with GLOBAL_LOAD_ASYNC_TO_LDS_B128 (ASYNCcnt).

#define KNN    20
#define NPTS   4096
#define NB     4
#define GN_EPS 1e-5f
#define LEAK   0.2f
#define CHUNKS 16

typedef __attribute__((ext_vector_type(16))) __bf16 bf16x16;
typedef __attribute__((ext_vector_type(8)))  float  floatx8;

__device__ __forceinline__ unsigned short f2bf_bits(float f) {
  union { float f; unsigned u; } x; x.f = f;
  unsigned r = x.u + 0x7FFFu + ((x.u >> 16) & 1u);   // RNE
  return (unsigned short)(r >> 16);
}
__device__ __forceinline__ float bf_bits2f(unsigned short h) {
  union { unsigned u; float f; } x; x.u = ((unsigned)h) << 16; return x.f;
}

// Fragment load: two contiguous 16B chunks of packed bf16.
__device__ __forceinline__ bf16x16 ld_frag(const unsigned short* p0, const unsigned short* p1) {
  union { bf16x16 v; uint4 q[2]; } u;
  u.q[0] = *(const uint4*)p0;
  u.q[1] = *(const uint4*)p1;
  return u.v;
}
// Store 8 f32 accumulators as one 16B bf16 chunk.
__device__ __forceinline__ void st_bf16x8(unsigned short* p, floatx8 a) {
  union { uint4 q; unsigned short s[8]; } u;
  #pragma unroll
  for (int i = 0; i < 8; ++i) u.s[i] = f2bf_bits(a[i]);
  *(uint4*)p = u.q;
}

#define WMMA_BF16(A, Bv, C) \
  __builtin_amdgcn_wmma_f32_16x16x32_bf16(false, (A), false, (Bv), (short)0, (C), false, false)

// ---------------- input packing: x -> bf16 rows -------------------------
__global__ void __launch_bounds__(256) pack_x_kernel(const float* __restrict__ x,
                                                     unsigned short* __restrict__ xd,
                                                     unsigned short* __restrict__ xg, int N) {
  int n = blockIdx.x * 256 + threadIdx.x;
  int b = blockIdx.y;
  if (n >= N) return;
  const float* xb = x + (size_t)b * 6 * N;
  unsigned short* d = xd + ((size_t)b * N + n) * 32;
  unsigned short* g = xg + ((size_t)b * N + n) * 16;
  #pragma unroll
  for (int c = 0; c < 32; ++c) d[c] = (c < 3) ? f2bf_bits(xb[(size_t)c * N + n]) : 0;
  #pragma unroll
  for (int c = 0; c < 16; ++c) g[c] = (c < 6) ? f2bf_bits(xb[(size_t)c * N + n]) : 0;
}

// ---------------- weight packing ----------------------------------------
__global__ void __launch_bounds__(256) wpack_plain_kernel(const float* __restrict__ W,
                                                          unsigned short* __restrict__ wb,
                                                          int O, int Cin, int Kpad) {
  int t = blockIdx.x * 256 + threadIdx.x;
  if (t >= O * Kpad) return;
  int o = t / Kpad, k = t - o * Kpad;
  wb[t] = (k < Cin) ? f2bf_bits(W[(size_t)o * Cin + k]) : 0;
}
// Edge conv: first CSP slots = Wa (nbr), next CSP = Wb - Wa (ctr).
__global__ void __launch_bounds__(256) wpack_edge_kernel(const float* __restrict__ W,
                                                         unsigned short* __restrict__ wb,
                                                         int O, int Cs, int CSP) {
  int Kpad = 2 * CSP;
  int t = blockIdx.x * 256 + threadIdx.x;
  if (t >= O * Kpad) return;
  int o = t / Kpad, k = t - o * Kpad;
  const float* wr = W + (size_t)o * 2 * Cs;
  float v = 0.f;
  if (k < CSP) { if (k < Cs) v = wr[k]; }
  else { int k2 = k - CSP; if (k2 < Cs) v = wr[Cs + k2] - wr[k2]; }
  wb[t] = f2bf_bits(v);
}

// ---------------- squared norms from packed rows ------------------------
__global__ void __launch_bounds__(256) sqnorm_kernel(const unsigned short* __restrict__ xt,
                                                     float* __restrict__ xx, int RS, int N) {
  int n = blockIdx.x * 256 + threadIdx.x;
  if (n >= N) return;
  const unsigned short* r = xt + (size_t)n * RS;
  float s = 0.f;
  for (int c = 0; c < RS; ++c) { float v = bf_bits2f(r[c]); s += v * v; }
  xx[n] = s;
}

// ---------------- pairwise -||xi-xj||^2: wave computes 16x64 ------------
template<int KSTEPS>
__global__ void __launch_bounds__(128) dist_kernel(const unsigned short* __restrict__ xt, // [N, 32*KSTEPS]
                                                   const float* __restrict__ xx,
                                                   float* __restrict__ pd, int N) {
  constexpr int RS = 32 * KSTEPS;
  const int lane = threadIdx.x & 31;
  const int wave = threadIdx.x >> 5;
  const int m = lane & 15, hi = (lane >> 4) & 1;
  const int ctpr = N >> 6;
  const int wid = blockIdx.x * 4 + wave;
  const int tr = wid / ctpr, tcg = wid - tr * ctpr;
  const int rowBase = tr << 4, colBase = tcg << 6;

  const unsigned short* arow = xt + (size_t)(rowBase + m) * RS;
  const floatx8 vzero = {0.f, 0.f, 0.f, 0.f, 0.f, 0.f, 0.f, 0.f};
  floatx8 acc[4];
  #pragma unroll
  for (int t = 0; t < 4; ++t) acc[t] = vzero;

  #pragma unroll
  for (int s = 0; s < KSTEPS; ++s) {
    const int ko = s * 32;
    bf16x16 a = ld_frag(arow + ko + hi * 16, arow + ko + hi * 16 + 8);
    bf16x16 bv[4];
    #pragma unroll
    for (int t = 0; t < 4; ++t) {
      const unsigned short* brow = xt + (size_t)(colBase + t * 16 + m) * RS + ko + hi * 16;
      bv[t] = ld_frag(brow, brow + 8);
    }
    #pragma unroll
    for (int t = 0; t < 4; ++t) acc[t] = WMMA_BF16(a, bv[t], acc[t]);
  }
  #pragma unroll
  for (int t = 0; t < 4; ++t) {
    int col = colBase + t * 16 + m;
    float xj = xx[col];
    #pragma unroll
    for (int v = 0; v < 8; ++v) {
      int row = rowBase + v + (hi << 3);
      pd[(size_t)row * N + col] = 2.f * acc[t][v] - xx[row] - xj;
    }
  }
}

// ---------------- top-k (k=20) per row ----------------------------------
// One wave per row. Row staged into LDS via GLOBAL_LOAD_ASYNC_TO_LDS_B128
// (ASYNCcnt, no VGPR round-trip), then 20 max-extraction passes with wave32
// butterflies. Tie -> lower index (lax.top_k ordering). srow is the only
// static LDS object in this kernel, so its LDS base offset is 0.
__global__ void __launch_bounds__(64) topk_kernel(const float* __restrict__ pd,
                                                  int* __restrict__ idx, int N) {
  __shared__ float srow[2][NPTS];
  const int lane = threadIdx.x & 31;
  const int w    = threadIdx.x >> 5;
  const int row  = blockIdx.x * 2 + w;
  const float* prow = pd + (size_t)row * N;

  // Async copy: 32 iterations x (32 lanes x 16B) = 16KB row.
  for (int i = 0; i < N / 128; ++i) {
    int e = (i * 32 + lane) * 4;                     // float index within row
    unsigned ldsoff = (unsigned)((w * NPTS + e) * 4);
    unsigned long long gaddr = (unsigned long long)(prow + e);
    asm volatile("global_load_async_to_lds_b128 %0, %1, off"
                 :: "v"(ldsoff), "v"(gaddr) : "memory");
  }
  asm volatile("s_wait_asynccnt 0x0" ::: "memory");

  for (int it = 0; it < KNN; ++it) {
    float bv = -3.402823466e38f; int bi = 0x7fffffff;
    for (int t = lane; t < N; t += 32) {
      float v = srow[w][t];
      if (v > bv || (v == bv && t < bi)) { bv = v; bi = t; }
    }
    #pragma unroll
    for (int msk = 16; msk >= 1; msk >>= 1) {
      float ov = __shfl_xor(bv, msk, 32);
      int   oi = __shfl_xor(bi, msk, 32);
      if (ov > bv || (ov == bv && oi < bi)) { bv = ov; bi = oi; }
    }
    if (lane == 0) { idx[row * KNN + it] = bi; srow[w][bi] = -3.402823466e38f; }
  }
}

// ---------------- gather conv: y = Wa·x[nbr] + (Wb-Wa)·x[ctr] ------------
template<int CSP>
__global__ void __launch_bounds__(128) gatherconv_kernel(
    const unsigned short* __restrict__ wb,   // [64, 2*CSP]
    const unsigned short* __restrict__ xt,   // [B, N, CSP]
    const int*   __restrict__ idx,           // [B, N*KNN]
    unsigned short* __restrict__ yT,         // [B, L, 64]
    int N) {
  constexpr int KSTEPS = CSP / 16;
  const int lane = threadIdx.x & 31, wave = threadIdx.x >> 5;
  const int m = lane & 15, hi = (lane >> 4) & 1;
  const int b = blockIdx.z;
  const int chBase = blockIdx.y << 4;
  const int colBase = (blockIdx.x * 4 + wave) << 6;
  const size_t L = (size_t)N * KNN;

  const unsigned short* wrow = wb + (size_t)(chBase + m) * (2 * CSP);
  const unsigned short* xb = xt + (size_t)b * N * CSP;
  const int* ib = idx + (size_t)b * L;

  const unsigned short* nbrRow[4]; const unsigned short* ctrRow[4];
  #pragma unroll
  for (int t = 0; t < 4; ++t) {
    int col = colBase + t * 16 + m;
    int n = col / KNN;
    int nb = ib[col];
    nbrRow[t] = xb + (size_t)nb * CSP;
    ctrRow[t] = xb + (size_t)n * CSP;
  }
  const floatx8 vzero = {0.f, 0.f, 0.f, 0.f, 0.f, 0.f, 0.f, 0.f};
  floatx8 acc[4];
  #pragma unroll
  for (int t = 0; t < 4; ++t) acc[t] = vzero;

  #pragma unroll
  for (int s = 0; s < KSTEPS; ++s) {
    const int ko = s * 32;
    bf16x16 a = ld_frag(wrow + ko + hi * 16, wrow + ko + hi * 16 + 8);
    const int gk = ko + hi * 16;     // half-fragment lies entirely in one row
    bf16x16 bv[4];
    #pragma unroll
    for (int t = 0; t < 4; ++t) {
      const unsigned short* rp = (gk < CSP) ? nbrRow[t] + gk : ctrRow[t] + (gk - CSP);
      bv[t] = ld_frag(rp, rp + 8);
    }
    #pragma unroll
    for (int t = 0; t < 4; ++t) acc[t] = WMMA_BF16(a, bv[t], acc[t]);
  }
  #pragma unroll
  for (int t = 0; t < 4; ++t) {
    size_t col = colBase + t * 16 + m;
    st_bf16x8(yT + ((size_t)b * L + col) * 64 + chBase + hi * 8, acc[t]);
  }
}

// ---------------- pure GEMM over N*k columns (conv2/conv4) --------------
template<int KSTEPS>
__global__ void __launch_bounds__(128) conv_nk_kernel(
    const unsigned short* __restrict__ wb,   // [64, 32*KSTEPS]
    const unsigned short* __restrict__ yin,  // [B, L, 32*KSTEPS] (activated)
    unsigned short* __restrict__ yout,       // [B, L, 64]
    size_t L) {
  constexpr int Cin = 32 * KSTEPS;
  const int lane = threadIdx.x & 31, wave = threadIdx.x >> 5;
  const int m = lane & 15, hi = (lane >> 4) & 1;
  const int b = blockIdx.z;
  const int chBase = blockIdx.y << 4;
  const size_t colBase = ((size_t)blockIdx.x * 4 + wave) << 6;
  const unsigned short* wrow = wb + (size_t)(chBase + m) * Cin;
  const unsigned short* rows[4];
  #pragma unroll
  for (int t = 0; t < 4; ++t)
    rows[t] = yin + ((size_t)b * L + colBase + t * 16 + m) * Cin;

  const floatx8 vzero = {0.f, 0.f, 0.f, 0.f, 0.f, 0.f, 0.f, 0.f};
  floatx8 acc[4];
  #pragma unroll
  for (int t = 0; t < 4; ++t) acc[t] = vzero;
  #pragma unroll
  for (int s = 0; s < KSTEPS; ++s) {
    const int ko = s * 32;
    bf16x16 a = ld_frag(wrow + ko + hi * 16, wrow + ko + hi * 16 + 8);
    bf16x16 bv[4];
    #pragma unroll
    for (int t = 0; t < 4; ++t)
      bv[t] = ld_frag(rows[t] + ko + hi * 16, rows[t] + ko + hi * 16 + 8);
    #pragma unroll
    for (int t = 0; t < 4; ++t) acc[t] = WMMA_BF16(a, bv[t], acc[t]);
  }
  #pragma unroll
  for (int t = 0; t < 4; ++t)
    st_bf16x8(yout + ((size_t)b * L + colBase + t * 16 + m) * 64 + chBase + hi * 8, acc[t]);
}

// ---------------- conv6: [x1;x2;x3] (192) -> 1024 over N ----------------
__global__ void __launch_bounds__(128) conv6_kernel(
    const unsigned short* __restrict__ wb,   // [1024, 192]
    const unsigned short* __restrict__ x1T, const unsigned short* __restrict__ x2T,
    const unsigned short* __restrict__ x3T,  // each [B, N, 64]
    unsigned short* __restrict__ y6T,        // [B, N, 1024]
    int N) {
  const int lane = threadIdx.x & 31, wave = threadIdx.x >> 5;
  const int m = lane & 15, hi = (lane >> 4) & 1;
  const int b = blockIdx.z;
  const int chBase = blockIdx.y << 4;
  const int colBase = (blockIdx.x * 4 + wave) << 6;
  const unsigned short* wrow = wb + (size_t)(chBase + m) * 192;
  const unsigned short* srcs[3] = {x1T, x2T, x3T};
  size_t rowOff[4];
  #pragma unroll
  for (int t = 0; t < 4; ++t) rowOff[t] = ((size_t)b * N + colBase + t * 16 + m) * 64;

  const floatx8 vzero = {0.f, 0.f, 0.f, 0.f, 0.f, 0.f, 0.f, 0.f};
  floatx8 acc[4];
  #pragma unroll
  for (int t = 0; t < 4; ++t) acc[t] = vzero;
  #pragma unroll
  for (int s = 0; s < 6; ++s) {
    bf16x16 a = ld_frag(wrow + s * 32 + hi * 16, wrow + s * 32 + hi * 16 + 8);
    const unsigned short* sp = srcs[s >> 1];
    const int off = ((s & 1) << 5) + hi * 16;
    bf16x16 bv[4];
    #pragma unroll
    for (int t = 0; t < 4; ++t)
      bv[t] = ld_frag(sp + rowOff[t] + off, sp + rowOff[t] + off + 8);
    #pragma unroll
    for (int t = 0; t < 4; ++t) acc[t] = WMMA_BF16(a, bv[t], acc[t]);
  }
  #pragma unroll
  for (int t = 0; t < 4; ++t)
    st_bf16x8(y6T + ((size_t)b * N + colBase + t * 16 + m) * 1024 + chBase + hi * 8, acc[t]);
}

// ---------------- conv7: [gvec;x1;x2;x3] (1216) -> 512 over N -----------
__global__ void __launch_bounds__(128) conv7_kernel(
    const unsigned short* __restrict__ wb,    // [512, 1216]
    const unsigned short* __restrict__ gvec,  // [B, 1024] (activated, broadcast)
    const unsigned short* __restrict__ x1T, const unsigned short* __restrict__ x2T,
    const unsigned short* __restrict__ x3T,
    unsigned short* __restrict__ y7T,         // [B, N, 512]
    int N) {
  const int lane = threadIdx.x & 31, wave = threadIdx.x >> 5;
  const int m = lane & 15, hi = (lane >> 4) & 1;
  const int b = blockIdx.z;
  const int chBase = blockIdx.y << 4;
  const int colBase = (blockIdx.x * 4 + wave) << 6;
  const unsigned short* wrow = wb + (size_t)(chBase + m) * 1216;
  const unsigned short* grow = gvec + (size_t)b * 1024;
  const unsigned short* srcs[3] = {x1T, x2T, x3T};
  size_t rowOff[4];
  #pragma unroll
  for (int t = 0; t < 4; ++t) rowOff[t] = ((size_t)b * N + colBase + t * 16 + m) * 64;

  const floatx8 vzero = {0.f, 0.f, 0.f, 0.f, 0.f, 0.f, 0.f, 0.f};
  floatx8 acc[4];
  #pragma unroll
  for (int t = 0; t < 4; ++t) acc[t] = vzero;
  #pragma unroll
  for (int s = 0; s < 38; ++s) {
    bf16x16 a = ld_frag(wrow + s * 32 + hi * 16, wrow + s * 32 + hi * 16 + 8);
    if (s < 32) {   // global descriptor: same B for all columns (broadcast)
      bf16x16 bv = ld_frag(grow + s * 32 + hi * 16, grow + s * 32 + hi * 16 + 8);
      #pragma unroll
      for (int t = 0; t < 4; ++t) acc[t] = WMMA_BF16(a, bv, acc[t]);
    } else {
      const unsigned short* sp = srcs[(s - 32) >> 1];
      const int off = (((s - 32) & 1) << 5) + hi * 16;
      bf16x16 bv[4];
      #pragma unroll
      for (int t = 0; t < 4; ++t)
        bv[t] = ld_frag(sp + rowOff[t] + off, sp + rowOff[t] + off + 8);
      #pragma unroll
      for (int t = 0; t < 4; ++t) acc[t] = WMMA_BF16(a, bv[t], acc[t]);
    }
  }
  #pragma unroll
  for (int t = 0; t < 4; ++t)
    st_bf16x8(y7T + ((size_t)b * N + colBase + t * 16 + m) * 512 + chBase + hi * 8, acc[t]);
}

// ---------------- GroupNorm stats: chunked partials (no atomics) --------
template<int C>
__global__ void __launch_bounds__(256) stats_part_kernel(const unsigned short* __restrict__ yT,
                                                         float* __restrict__ part, // [B,32,CHUNKS,2]
                                                         size_t Lc) {
  constexpr int cpg = C / 32;
  constexpr int colsPerIter = 256 / cpg;
  const int chunk = blockIdx.x, g = blockIdx.y, b = blockIdx.z;
  const size_t colsPerChunk = Lc / CHUNKS;
  const int e = threadIdx.x % cpg;
  const int cofs = threadIdx.x / cpg;
  const unsigned short* base = yT + (size_t)b * Lc * C + (size_t)g * cpg + e;
  float s = 0.f, ss = 0.f;
  const size_t cend = (size_t)(chunk + 1) * colsPerChunk;
  for (size_t col = (size_t)chunk * colsPerChunk + cofs; col < cend; col += colsPerIter) {
    float v = bf_bits2f(base[col * C]);
    s += v; ss += v * v;
  }
  __shared__ float rs[256], rq[256];
  rs[threadIdx.x] = s; rq[threadIdx.x] = ss;
  __syncthreads();
  for (int o = 128; o > 0; o >>= 1) {
    if (threadIdx.x < o) { rs[threadIdx.x] += rs[threadIdx.x + o]; rq[threadIdx.x] += rq[threadIdx.x + o]; }
    __syncthreads();
  }
  if (threadIdx.x == 0) {
    size_t p = (((size_t)b * 32 + g) * CHUNKS + chunk) * 2;
    part[p] = rs[0]; part[p + 1] = rq[0];
  }
}

__global__ void __launch_bounds__(128) stats_final_kernel(const float* __restrict__ part,
                                                          float* __restrict__ S, float invM) {
  int t = threadIdx.x;                      // B*32 == 128
  if (t >= NB * 32) return;
  float s = 0.f, ss = 0.f;
  for (int c = 0; c < CHUNKS; ++c) { s += part[(t * CHUNKS + c) * 2]; ss += part[(t * CHUNKS + c) * 2 + 1]; }
  float mu = s * invM;
  float var = ss * invM - mu * mu;
  S[t * 2] = mu; S[t * 2 + 1] = rsqrtf(var + GN_EPS);
}

// ---------------- in-place GN + LeakyReLU -------------------------------
template<int C>
__global__ void __launch_bounds__(256) gn_apply_kernel(unsigned short* __restrict__ yT,
                                                       const float* __restrict__ S,
                                                       const float* __restrict__ gam,
                                                       const float* __restrict__ bet, size_t Lc) {
  constexpr int cpg = C / 32;
  size_t t = (size_t)blockIdx.x * 256 + threadIdx.x;
  const int b = blockIdx.y;
  if (t >= Lc * C) return;
  int c = (int)(t & (C - 1));
  int g = c / cpg;
  float mu = S[(b * 32 + g) * 2], r = S[(b * 32 + g) * 2 + 1];
  unsigned short* p = yT + (size_t)b * Lc * C + t;
  float z = gam[c] * (bf_bits2f(*p) - mu) * r + bet[c];
  z = z > 0.f ? z : LEAK * z;
  *p = f2bf_bits(z);
}

// ---------------- max over k -> packed x rows ---------------------------
__global__ void __launch_bounds__(256) maxk_kernel(const unsigned short* __restrict__ yT, // [B,L,64] act'd
                                                   unsigned short* __restrict__ xT,       // [B,N,64]
                                                   int N) {
  int t = blockIdx.x * 256 + threadIdx.x;
  int b = blockIdx.y;
  if (t >= N * 64) return;
  int n = t >> 6, c = t & 63;
  const unsigned short* p = yT + ((size_t)b * N + n) * (size_t)KNN * 64 + c;
  float mx = -3.402823466e38f;
  #pragma unroll
  for (int j = 0; j < KNN; ++j) mx = fmaxf(mx, bf_bits2f(p[j * 64]));
  xT[((size_t)b * N + n) * 64 + c] = f2bf_bits(mx);
}

// ---------------- global max over N (conv6 output) ----------------------
__global__ void __launch_bounds__(256) gmax_part_kernel(const unsigned short* __restrict__ y6T,
                                                        float* __restrict__ gpart, // [B,16,1024]
                                                        int N) {
  const int chunk = blockIdx.x, b = blockIdx.z;
  const int c = blockIdx.y * 256 + threadIdx.x;
  const int npc = N / 16;
  const unsigned short* p = y6T + ((size_t)b * N + (size_t)chunk * npc) * 1024 + c;
  float mx = -3.402823466e38f;
  for (int n = 0; n < npc; ++n) mx = fmaxf(mx, bf_bits2f(p[(size_t)n * 1024]));
  gpart[((size_t)b * 16 + chunk) * 1024 + c] = mx;
}
__global__ void __launch_bounds__(256) gmax_final_kernel(const float* __restrict__ gpart,
                                                         unsigned short* __restrict__ gvec) {
  int t = blockIdx.x * 256 + threadIdx.x;
  if (t >= NB * 1024) return;
  int b = t >> 10, c = t & 1023;
  float mx = -3.402823466e38f;
  for (int ch = 0; ch < 16; ++ch) mx = fmaxf(mx, gpart[((size_t)b * 16 + ch) * 1024 + c]);
  gvec[t] = f2bf_bits(mx);
}

// ---------------- final: GN+LeakyReLU -> f32 [B,512,N] ------------------
__global__ void __launch_bounds__(256) final_kernel(const unsigned short* __restrict__ y7T, // [B,N,512]
    const float* __restrict__ S, const float* __restrict__ gam, const float* __restrict__ bet,
    float* __restrict__ out, int N) {
  size_t t = (size_t)blockIdx.x * 256 + threadIdx.x;   // enumerates (c, n)
  int b = blockIdx.y;
  if (t >= (size_t)512 * N) return;
  int c = (int)(t / N), n = (int)(t - (size_t)c * N);
  float v = bf_bits2f(y7T[((size_t)b * N + n) * 512 + c]);
  int g = c >> 4;                                      // cpg = 16
  float mu = S[(b * 32 + g) * 2], r = S[(b * 32 + g) * 2 + 1];
  float z = gam[c] * (v - mu) * r + bet[c];
  out[(size_t)b * 512 * N + t] = z > 0.f ? z : LEAK * z;
}

// =======================================================================
extern "C" void kernel_launch(void* const* d_in, const int* in_sizes, int n_in,
                              void* d_out, int out_size, void* d_ws, size_t ws_size,
                              hipStream_t stream) {
  (void)in_sizes; (void)n_in; (void)out_size; (void)ws_size;
  const int B = NB, N = NPTS;
  const size_t L = (size_t)N * KNN;

  const float* x = (const float*)d_in[0];
  const float* Wp[8]; const float* gp[8]; const float* bp[8];
  for (int j = 1; j <= 7; ++j) {
    Wp[j] = (const float*)d_in[1 + 3 * (j - 1)];
    gp[j] = (const float*)d_in[2 + 3 * (j - 1)];
    bp[j] = (const float*)d_in[3 + 3 * (j - 1)];
  }

  char* ws = (char*)d_ws;
  size_t off = 0;
  auto alloc = [&](size_t bytes) -> char* {
    char* p = ws + off; off = (off + bytes + 255) & ~(size_t)255; return p;
  };
  float* pd   = (float*)alloc((size_t)N * N * 4);              // 64 MB, per-batch reuse
  float* xx   = (float*)alloc((size_t)N * 4);
  unsigned short* xd  = (unsigned short*)alloc((size_t)B * N * 32 * 2);
  unsigned short* xg  = (unsigned short*)alloc((size_t)B * N * 16 * 2);
  int* idx1 = (int*)alloc((size_t)B * L * 4);
  int* idx2 = (int*)alloc((size_t)B * L * 4);
  int* idx3 = (int*)alloc((size_t)B * L * 4);
  unsigned short* x1T = (unsigned short*)alloc((size_t)B * N * 64 * 2);
  unsigned short* x2T = (unsigned short*)alloc((size_t)B * N * 64 * 2);
  unsigned short* x3T = (unsigned short*)alloc((size_t)B * N * 64 * 2);
  unsigned short* gvec = (unsigned short*)alloc((size_t)B * 1024 * 2);
  float* gpart = (float*)alloc((size_t)B * 16 * 1024 * 4);
  float* S     = (float*)alloc((size_t)B * 32 * 2 * 4);
  float* part  = (float*)alloc((size_t)B * 32 * CHUNKS * 2 * 4);
  unsigned short* wb1 = (unsigned short*)alloc(64 * 32 * 2);
  unsigned short* wb2 = (unsigned short*)alloc(64 * 64 * 2);
  unsigned short* wb3 = (unsigned short*)alloc(64 * 128 * 2);
  unsigned short* wb4 = (unsigned short*)alloc(64 * 64 * 2);
  unsigned short* wb5 = (unsigned short*)alloc(64 * 128 * 2);
  unsigned short* wb6 = (unsigned short*)alloc(1024 * 192 * 2);
  unsigned short* wb7 = (unsigned short*)alloc((size_t)512 * 1216 * 2);
  unsigned short* yA = (unsigned short*)alloc((size_t)B * L * 64 * 2);  // also y6T [B,N,1024]
  unsigned short* yB = (unsigned short*)alloc((size_t)B * L * 64 * 2);  // also y7T [B,N,512]

  const unsigned distBlocks = (unsigned)((N / 16) * (N / 64) / 4);   // 4096
  const unsigned nkBlocks   = (unsigned)(L / 64 / 4);               // 320
  const unsigned nBlocks    = (unsigned)(N / 64 / 4);               // 16

  // ---- packing ----
  pack_x_kernel<<<dim3(N / 256, B), 256, 0, stream>>>(x, xd, xg, N);
  wpack_edge_kernel<<<(64 * 32 + 255) / 256, 256, 0, stream>>>(Wp[1], wb1, 64, 6, 16);
  wpack_plain_kernel<<<(64 * 64 + 255) / 256, 256, 0, stream>>>(Wp[2], wb2, 64, 64, 64);
  wpack_edge_kernel<<<(64 * 128 + 255) / 256, 256, 0, stream>>>(Wp[3], wb3, 64, 64, 64);
  wpack_plain_kernel<<<(64 * 64 + 255) / 256, 256, 0, stream>>>(Wp[4], wb4, 64, 64, 64);
  wpack_edge_kernel<<<(64 * 128 + 255) / 256, 256, 0, stream>>>(Wp[5], wb5, 64, 64, 64);
  wpack_plain_kernel<<<(1024 * 192 + 255) / 256, 256, 0, stream>>>(Wp[6], wb6, 1024, 192, 192);
  wpack_plain_kernel<<<(512 * 1216 + 255) / 256, 256, 0, stream>>>(Wp[7], wb7, 512, 1216, 1216);

  // ---- Block 1 ----
  for (int b = 0; b < B; ++b) {
    const unsigned short* xb = xd + (size_t)b * N * 32;
    sqnorm_kernel<<<N / 256, 256, 0, stream>>>(xb, xx, 32, N);
    dist_kernel<1><<<distBlocks, 128, 0, stream>>>(xb, xx, pd, N);
    topk_kernel<<<N / 2, 64, 0, stream>>>(pd, idx1 + (size_t)b * L, N);
  }
  gatherconv_kernel<16><<<dim3(nkBlocks, 4, B), 128, 0, stream>>>(wb1, xg, idx1, yA, N);
  stats_part_kernel<64><<<dim3(CHUNKS, 32, B), 256, 0, stream>>>(yA, part, L);
  stats_final_kernel<<<1, 128, 0, stream>>>(part, S, 1.f / (float)(2 * L));
  gn_apply_kernel<64><<<dim3((unsigned)(L * 64 / 256), B), 256, 0, stream>>>(yA, S, gp[1], bp[1], L);
  conv_nk_kernel<2><<<dim3(nkBlocks, 4, B), 128, 0, stream>>>(wb2, yA, yB, L);
  stats_part_kernel<64><<<dim3(CHUNKS, 32, B), 256, 0, stream>>>(yB, part, L);
  stats_final_kernel<<<1, 128, 0, stream>>>(part, S, 1.f / (float)(2 * L));
  gn_apply_kernel<64><<<dim3((unsigned)(L * 64 / 256), B), 256, 0, stream>>>(yB, S, gp[2], bp[2], L);
  maxk_kernel<<<dim3(N * 64 / 256, B), 256, 0, stream>>>(yB, x1T, N);

  // ---- Block 2 ----
  for (int b = 0; b < B; ++b) {
    const unsigned short* xb = x1T + (size_t)b * N * 64;
    sqnorm_kernel<<<N / 256, 256, 0, stream>>>(xb, xx, 64, N);
    dist_kernel<2><<<distBlocks, 128, 0, stream>>>(xb, xx, pd, N);
    topk_kernel<<<N / 2, 64, 0, stream>>>(pd, idx2 + (size_t)b * L, N);
  }
  gatherconv_kernel<64><<<dim3(nkBlocks, 4, B), 128, 0, stream>>>(wb3, x1T, idx2, yA, N);
  stats_part_kernel<64><<<dim3(CHUNKS, 32, B), 256, 0, stream>>>(yA, part, L);
  stats_final_kernel<<<1, 128, 0, stream>>>(part, S, 1.f / (float)(2 * L));
  gn_apply_kernel<64><<<dim3((unsigned)(L * 64 / 256), B), 256, 0, stream>>>(yA, S, gp[3], bp[3], L);
  conv_nk_kernel<2><<<dim3(nkBlocks, 4, B), 128, 0, stream>>>(wb4, yA, yB, L);
  stats_part_kernel<64><<<dim3(CHUNKS, 32, B), 256, 0, stream>>>(yB, part, L);
  stats_final_kernel<<<1, 128, 0, stream>>>(part, S, 1.f / (float)(2 * L));
  gn_apply_kernel<64><<<dim3((unsigned)(L * 64 / 256), B), 256, 0, stream>>>(yB, S, gp[4], bp[4], L);
  maxk_kernel<<<dim3(N * 64 / 256, B), 256, 0, stream>>>(yB, x2T, N);

  // ---- Block 3 ----
  for (int b = 0; b < B; ++b) {
    const unsigned short* xb = x2T + (size_t)b * N * 64;
    sqnorm_kernel<<<N / 256, 256, 0, stream>>>(xb, xx, 64, N);
    dist_kernel<2><<<distBlocks, 128, 0, stream>>>(xb, xx, pd, N);
    topk_kernel<<<N / 2, 64, 0, stream>>>(pd, idx3 + (size_t)b * L, N);
  }
  gatherconv_kernel<64><<<dim3(nkBlocks, 4, B), 128, 0, stream>>>(wb5, x2T, idx3, yA, N);
  stats_part_kernel<64><<<dim3(CHUNKS, 32, B), 256, 0, stream>>>(yA, part, L);
  stats_final_kernel<<<1, 128, 0, stream>>>(part, S, 1.f / (float)(2 * L));
  gn_apply_kernel<64><<<dim3((unsigned)(L * 64 / 256), B), 256, 0, stream>>>(yA, S, gp[5], bp[5], L);
  maxk_kernel<<<dim3(N * 64 / 256, B), 256, 0, stream>>>(yA, x3T, N);

  // ---- conv6 + global max ----
  conv6_kernel<<<dim3(nBlocks, 64, B), 128, 0, stream>>>(wb6, x1T, x2T, x3T, yA, N);
  stats_part_kernel<1024><<<dim3(CHUNKS, 32, B), 256, 0, stream>>>(yA, part, (size_t)N);
  stats_final_kernel<<<1, 128, 0, stream>>>(part, S, 1.f / (float)(32 * N));
  gn_apply_kernel<1024><<<dim3((unsigned)((size_t)N * 1024 / 256), B), 256, 0, stream>>>(yA, S, gp[6], bp[6], (size_t)N);
  gmax_part_kernel<<<dim3(16, 4, B), 256, 0, stream>>>(yA, gpart, N);
  gmax_final_kernel<<<(NB * 1024 + 255) / 256, 256, 0, stream>>>(gpart, gvec);

  // ---- conv7 + final ----
  conv7_kernel<<<dim3(nBlocks, 32, B), 128, 0, stream>>>(wb7, gvec, x1T, x2T, x3T, yB, N);
  stats_part_kernel<512><<<dim3(CHUNKS, 32, B), 256, 0, stream>>>(yB, part, (size_t)N);
  stats_final_kernel<<<1, 128, 0, stream>>>(part, S, 1.f / (float)(16 * N));
  final_kernel<<<dim3((unsigned)((size_t)512 * N / 256), B), 256, 0, stream>>>(yB, S, gp[7], bp[7], (float*)d_out, N);
}
